// re_unit_48103633715562
// MI455X (gfx1250) — matrix operand order repaired and checked by reading, not compiled, not run on hardware
//
#include <hip/hip_runtime.h>
#include <stdint.h>

// Problem constants (from reference): L=128, B=4, H=768, R=24
#define Hd   768
#define Lseq 128
#define Bb   4
#define Rr   24
#define MROWS (Lseq * Bb)          // 512 rows for stage-A GEMMs
#define LN_EPS 1e-5f

typedef __attribute__((ext_vector_type(16))) __bf16 v16bf;
typedef __attribute__((ext_vector_type(8)))  float  v8f;

union Frag {                // 32 bytes = 16 bf16 = one WMMA A/B operand
    uint4 q[2];
    v16bf v;
};

__device__ __forceinline__ uint16_t f32_to_bf16(float f) {
    union { float f; uint32_t u; } c; c.f = f;
    uint32_t u = c.u;
    u += 0x7FFFu + ((u >> 16) & 1u);      // round to nearest even
    return (uint16_t)(u >> 16);
}

// ---------------------------------------------------------------- converts
__global__ void cvt_bf16_kernel(const float* __restrict__ src,
                                uint16_t* __restrict__ dst, int n) {
    int i = blockIdx.x * blockDim.x + threadIdx.x;
    if (i < n) dst[i] = f32_to_bf16(src[i]);
}

// rel_w (24x768) -> bf16 padded to 32x768 (rows 24..31 zero for the N-tile pad)
__global__ void cvt_relw_pad_kernel(const float* __restrict__ src,
                                    uint16_t* __restrict__ dst) {
    int i = blockIdx.x * blockDim.x + threadIdx.x;     // 0..32*768-1
    if (i < 32 * Hd) dst[i] = (i < Rr * Hd) ? f32_to_bf16(src[i]) : (uint16_t)0;
}

// ---------------------------------------------------------------- stage A GEMMs
// One wave = one 16x64 output strip (1 M-tile x 4 N-tiles, A fragments reused 4x).
// D[m,n] = sum_h A[m,h] * Wrow[n][h]
// A frag (16x32 bf16): lanes 0-15 row=lane, K 0..15; lanes 16-31 K 16..31
// B frag (32x16 bf16): lanes 0-15 col=lane, K 0..15; lanes 16-31 col=lane-16, K 16..31
// mode 0: p1 = h_re @ w1^T        (w1[n,h] = hid_w[n, h])
// mode 1: p2 = h_re @ w2^T        (w2[n,h] = hid_w[n, H+h])
// mode 2: t  = tanh(h_share@w_sh^T + h_re@w_re^T + r_b)
__global__ __launch_bounds__(32)
void stageA_gemm_kernel(const uint16_t* __restrict__ hre_bf,
                        const uint16_t* __restrict__ hsh_bf,
                        const uint16_t* __restrict__ rw_bf,   // 768 x 1536
                        const uint16_t* __restrict__ hid_bf,  // 768 x 2304
                        const float*    __restrict__ r_b,
                        float* __restrict__ p1, float* __restrict__ p2,
                        float* __restrict__ tbuf) {
    const int tile = blockIdx.x;               // 0..(32*12-1)
    const int mode = blockIdx.y;               // 0..2
    const int ng = tile % 12, mt = tile / 12;
    const int m0 = mt * 16, n0 = ng * 64;
    const int lane = threadIdx.x;
    const int aRow = m0 + (lane & 15);
    const int aO   = (lane >> 4) * 16;
    const int bL   = lane & 15;
    const int bO   = (lane >> 4) * 16;

    v8f acc[4] = {{}, {}, {}, {}};
    if (mode == 2) {
        for (int kc = 0; kc < 24; ++kc) {
            Frag a1, a2;
            const uint16_t* ap1 = hsh_bf + aRow * Hd + kc * 32 + aO;
            a1.q[0] = *(const uint4*)(ap1);
            a1.q[1] = *(const uint4*)(ap1 + 8);
            const uint16_t* ap2 = hre_bf + aRow * Hd + kc * 32 + aO;
            a2.q[0] = *(const uint4*)(ap2);
            a2.q[1] = *(const uint4*)(ap2 + 8);
#pragma unroll
            for (int tn = 0; tn < 4; ++tn) {
                const int col = n0 + tn * 16 + bL;
                Frag b1, b2;
                const uint16_t* bp1 = rw_bf + col * (2 * Hd) + kc * 32 + bO;
                b1.q[0] = *(const uint4*)(bp1);
                b1.q[1] = *(const uint4*)(bp1 + 8);
                const uint16_t* bp2 = rw_bf + col * (2 * Hd) + Hd + kc * 32 + bO;
                b2.q[0] = *(const uint4*)(bp2);
                b2.q[1] = *(const uint4*)(bp2 + 8);
                acc[tn] = __builtin_amdgcn_wmma_f32_16x16x32_bf16(false, a1.v, false, b1.v,
                                                                  (short)0, acc[tn], false, false);
                acc[tn] = __builtin_amdgcn_wmma_f32_16x16x32_bf16(false, a2.v, false, b2.v,
                                                                  (short)0, acc[tn], false, false);
            }
        }
    } else {
        const int wsel = mode;                 // 0 -> w1, 1 -> w2
        for (int kc = 0; kc < 24; ++kc) {
            Frag a;
            const uint16_t* ap = hre_bf + aRow * Hd + kc * 32 + aO;
            a.q[0] = *(const uint4*)(ap);
            a.q[1] = *(const uint4*)(ap + 8);
#pragma unroll
            for (int tn = 0; tn < 4; ++tn) {
                const int col = n0 + tn * 16 + bL;
                Frag b;
                const uint16_t* bp = hid_bf + col * (3 * Hd) + wsel * Hd + kc * 32 + bO;
                b.q[0] = *(const uint4*)(bp);
                b.q[1] = *(const uint4*)(bp + 8);
                acc[tn] = __builtin_amdgcn_wmma_f32_16x16x32_bf16(false, a.v, false, b.v,
                                                                  (short)0, acc[tn], false, false);
            }
        }
    }

    // C/D layout: lanes 0-15: N=lane, M=v; lanes 16-31: N=lane-16, M=v+8
#pragma unroll
    for (int tn = 0; tn < 4; ++tn) {
        const int col = n0 + tn * 16 + bL;
#pragma unroll
        for (int v = 0; v < 8; ++v) {
            int row = m0 + v + ((lane >> 4) * 8);
            float d = acc[tn][v];
            if (mode == 0)      p1[row * Hd + col] = d;
            else if (mode == 1) p2[row * Hd + col] = d;
            else                tbuf[row * Hd + col] = tanhf(d + r_b[col]);
        }
    }
}

// ------------------------------------------------- g = max over l of tanh-path
__global__ void gmax_kernel(const float* __restrict__ tbuf, float* __restrict__ g) {
    int t = blockIdx.x * blockDim.x + threadIdx.x;     // 0..(4*768-1)
    if (t >= Bb * Hd) return;
    int b = t / Hd, k = t % Hd;
    float m = -3.402823466e38f;
    for (int l = 0; l < Lseq; ++l) {
        float v = tbuf[(l * Bb + b) * Hd + k];
        m = fmaxf(m, v);
    }
    g[b * Hd + k] = m;
}

// ------------------------------------------------- pg = g @ w3^T + hid_b (tiny, f32)
__global__ void pg_kernel(const float* __restrict__ g, const float* __restrict__ hid_w,
                          const float* __restrict__ hid_b, float* __restrict__ pg) {
    int t = blockIdx.x * blockDim.x + threadIdx.x;     // 0..(4*768-1)
    if (t >= Bb * Hd) return;
    int b = t / Hd, k = t % Hd;
    const float* gv = g + b * Hd;
    const float* wv = hid_w + k * (3 * Hd) + 2 * Hd;   // w3[k,h]
    float acc = hid_b[k];
    for (int h = 0; h < Hd; ++h) acc += gv[h] * wv[h];
    pg[b * Hd + k] = acc;
}

// ---------------------------------------------------------------- fused epilogue
// One wave owns 16 output rows m = tile*16 .. +15 (i constant per tile).
// Phase 1: build z rows (p1+p2+pg, float4 loads), wave-reduce LN stats, elu,
//          pack bf16 x4 -> ds_store_b64 into per-wave LDS A-buffer.
// Phase 2: 24 K-chunks x 2 N-tiles of v_wmma_f32_16x16x32_bf16 against rel_w in LDS.
// Phase 3: sigmoid + mask, scatter 16x24 outputs.
__global__ __launch_bounds__(128)
void fused_out_kernel(const float* __restrict__ p1, const float* __restrict__ p2,
                      const float* __restrict__ pg,
                      const float* __restrict__ ln_g, const float* __restrict__ ln_b,
                      const uint16_t* __restrict__ relw_bf,   // 32 x 768 bf16 (padded)
                      const float* __restrict__ rel_b,
                      const float* __restrict__ mask,         // (L,B)
                      float* __restrict__ out) {
    extern __shared__ char smem[];
    uint16_t* relw_lds = (uint16_t*)smem;                       // 32*768 bf16 = 48 KB
    uint16_t* abuf     = (uint16_t*)(smem + 32 * Hd * 2);       // 4 waves * 16*768 bf16

    {   // vectorized rel_w fill: 3072 uint4 over 128 threads
        uint4* d4 = (uint4*)relw_lds;
        const uint4* s4 = (const uint4*)relw_bf;
        for (int idx = threadIdx.x; idx < (32 * Hd * 2) / 16; idx += blockDim.x)
            d4[idx] = s4[idx];
    }
    __syncthreads();

    const int wave = threadIdx.x >> 5;
    const int lane = threadIdx.x & 31;
    const int tile = blockIdx.x * 4 + wave;
    const int m0   = tile * 16;
    const int i    = m0 >> 9;                  // m = i*512 + j*4 + b
    uint16_t* arow_base = abuf + wave * 16 * Hd;

    // hoisted LN params: lane handles k = 4*(c*32+lane) .. +3
    float4 lg[6], lb[6];
    {
        const float4* LG = (const float4*)ln_g;
        const float4* LB = (const float4*)ln_b;
#pragma unroll
        for (int c = 0; c < 6; ++c) {
            lg[c] = LG[c * 32 + lane];
            lb[c] = LB[c * 32 + lane];
        }
    }

    // ---- phase 1: 16 z-rows, LN + elu, store bf16 to LDS
    for (int r = 0; r < 16; ++r) {
        int m = m0 + r;
        int j = (m >> 2) & 127;
        int b = m & 3;
        const float4* P1 = (const float4*)(p1 + (i * Bb + b) * Hd);
        const float4* P2 = (const float4*)(p2 + (j * Bb + b) * Hd);
        const float4* PG = (const float4*)(pg + b * Hd);
        float4 zr[6];
        float s = 0.f, s2 = 0.f;
#pragma unroll
        for (int c = 0; c < 6; ++c) {
            int idx = c * 32 + lane;
            float4 a = P1[idx], q = P2[idx], g4 = PG[idx];
            float4 z;
            z.x = a.x + q.x + g4.x;
            z.y = a.y + q.y + g4.y;
            z.z = a.z + q.z + g4.z;
            z.w = a.w + q.w + g4.w;
            zr[c] = z;
            s  += z.x + z.y + z.z + z.w;
            s2 += z.x * z.x + z.y * z.y + z.z * z.z + z.w * z.w;
        }
        for (int o = 16; o >= 1; o >>= 1) {    // wave32 tree reduce
            s  += __shfl_xor(s,  o, 32);
            s2 += __shfl_xor(s2, o, 32);
        }
        const float inv = 1.f / (float)Hd;
        float mu  = s * inv;
        float var = s2 * inv - mu * mu;
        float rs  = rsqrtf(var + LN_EPS);
        uint2* dst = (uint2*)(arow_base + r * Hd);   // 4 bf16 per uint2
#pragma unroll
        for (int c = 0; c < 6; ++c) {
            int idx = c * 32 + lane;
            float4 z = zr[c];
            float yx = (z.x - mu) * rs * lg[c].x + lb[c].x;
            float yy = (z.y - mu) * rs * lg[c].y + lb[c].y;
            float yz = (z.z - mu) * rs * lg[c].z + lb[c].z;
            float yw = (z.w - mu) * rs * lg[c].w + lb[c].w;
            yx = (yx > 0.f) ? yx : (__expf(yx) - 1.f);   // elu
            yy = (yy > 0.f) ? yy : (__expf(yy) - 1.f);
            yz = (yz > 0.f) ? yz : (__expf(yz) - 1.f);
            yw = (yw > 0.f) ? yw : (__expf(yw) - 1.f);
            uint32_t lo = (uint32_t)f32_to_bf16(yx) | ((uint32_t)f32_to_bf16(yy) << 16);
            uint32_t hi = (uint32_t)f32_to_bf16(yz) | ((uint32_t)f32_to_bf16(yw) << 16);
            dst[idx] = make_uint2(lo, hi);
        }
    }
    // Same-wave LDS ops are in-order; no workgroup barrier needed.

    // ---- phase 2: z(16x768) @ rel_w^T(768x32) via WMMA
    v8f acc0 = {}, acc1 = {};
    const int aRow = lane & 15;
    const int aO   = (lane >> 4) * 16;
    const int bCol = lane & 15;
    const int bO   = (lane >> 4) * 16;
    for (int kc = 0; kc < 24; ++kc) {
        Frag a, b0, b1;
        const uint16_t* ap = arow_base + aRow * Hd + kc * 32 + aO;
        a.q[0] = *(const uint4*)(ap);
        a.q[1] = *(const uint4*)(ap + 8);
        const uint16_t* bp0 = relw_lds + bCol * Hd + kc * 32 + bO;
        b0.q[0] = *(const uint4*)(bp0);
        b0.q[1] = *(const uint4*)(bp0 + 8);
        const uint16_t* bp1 = relw_lds + (16 + bCol) * Hd + kc * 32 + bO;  // rows 24..31 zero
        b1.q[0] = *(const uint4*)(bp1);
        b1.q[1] = *(const uint4*)(bp1 + 8);
        acc0 = __builtin_amdgcn_wmma_f32_16x16x32_bf16(false, a.v, false, b0.v,
                                                       (short)0, acc0, false, false);
        acc1 = __builtin_amdgcn_wmma_f32_16x16x32_bf16(false, a.v, false, b1.v,
                                                       (short)0, acc1, false, false);
    }

    // ---- phase 3: sigmoid + mask, store (r = n and r = 16+n for n<8)
    const int n = lane & 15;
#pragma unroll
    for (int v = 0; v < 8; ++v) {
        int Mr = v + ((lane >> 4) * 8);
        int m  = m0 + Mr;
        int j  = (m >> 2) & 127;
        int b  = m & 3;
        float mf = mask[i * Bb + b] * mask[j * Bb + b];
        float x0 = acc0[v] + rel_b[n];
        out[m * Rr + n] = mf / (1.f + __expf(-x0));
        if (n < 8) {
            float x1 = acc1[v] + rel_b[16 + n];
            out[m * Rr + 16 + n] = mf / (1.f + __expf(-x1));
        }
    }
}

// ---------------------------------------------------------------- launch
extern "C" void kernel_launch(void* const* d_in, const int* in_sizes, int n_in,
                              void* d_out, int out_size, void* d_ws, size_t ws_size,
                              hipStream_t stream) {
    const float* h_re    = (const float*)d_in[0];
    const float* h_share = (const float*)d_in[1];
    const float* mask    = (const float*)d_in[2];
    const float* r_w     = (const float*)d_in[3];
    const float* r_b     = (const float*)d_in[4];
    const float* hid_w   = (const float*)d_in[5];
    const float* hid_b   = (const float*)d_in[6];
    const float* ln_g    = (const float*)d_in[7];
    const float* ln_b    = (const float*)d_in[8];
    const float* rel_w   = (const float*)d_in[9];
    const float* rel_b   = (const float*)d_in[10];
    float* out = (float*)d_out;

    // workspace carve-up (~12.3 MB total)
    char* ws = (char*)d_ws;
    size_t off = 0;
    auto carve = [&](size_t bytes) -> char* {
        char* p = ws + off;
        off = (off + bytes + 255) & ~(size_t)255;
        return p;
    };
    uint16_t* hre_bf  = (uint16_t*)carve((size_t)MROWS * Hd * 2);
    uint16_t* hsh_bf  = (uint16_t*)carve((size_t)MROWS * Hd * 2);
    uint16_t* rw_bf   = (uint16_t*)carve((size_t)Hd * 2 * Hd * 2);
    uint16_t* hid_bf  = (uint16_t*)carve((size_t)Hd * 3 * Hd * 2);
    uint16_t* relw_bf = (uint16_t*)carve((size_t)32 * Hd * 2);
    float*    p1      = (float*)carve((size_t)MROWS * Hd * 4);
    float*    p2      = (float*)carve((size_t)MROWS * Hd * 4);
    float*    tbuf    = (float*)carve((size_t)MROWS * Hd * 4);
    float*    g       = (float*)carve((size_t)Bb * Hd * 4);
    float*    pg      = (float*)carve((size_t)Bb * Hd * 4);
    (void)ws_size; (void)in_sizes; (void)n_in; (void)out_size;

    const int T = 256;
    auto blks = [](int n, int t) { return (n + t - 1) / t; };

    cvt_bf16_kernel<<<blks(MROWS * Hd, T), T, 0, stream>>>(h_re, hre_bf, MROWS * Hd);
    cvt_bf16_kernel<<<blks(MROWS * Hd, T), T, 0, stream>>>(h_share, hsh_bf, MROWS * Hd);
    cvt_bf16_kernel<<<blks(Hd * 2 * Hd, T), T, 0, stream>>>(r_w, rw_bf, Hd * 2 * Hd);
    cvt_bf16_kernel<<<blks(Hd * 3 * Hd, T), T, 0, stream>>>(hid_w, hid_bf, Hd * 3 * Hd);
    cvt_relw_pad_kernel<<<blks(32 * Hd, T), T, 0, stream>>>(rel_w, relw_bf);

    dim3 gA((MROWS / 16) * (Hd / 64), 3);          // 32 M-tiles x 12 N-groups x 3 modes
    stageA_gemm_kernel<<<gA, 32, 0, stream>>>(hre_bf, hsh_bf, rw_bf, hid_bf,
                                              r_b, p1, p2, tbuf);

    gmax_kernel<<<blks(Bb * Hd, T), T, 0, stream>>>(tbuf, g);
    pg_kernel<<<blks(Bb * Hd, T), T, 0, stream>>>(g, hid_w, hid_b, pg);

    const int nTiles = (Lseq * Lseq * Bb) / 16;    // 4096
    const size_t shmem = (size_t)(32 * Hd * 2) + (size_t)(4 * 16 * Hd * 2);  // 144 KB
    fused_out_kernel<<<nTiles / 4, 128, shmem, stream>>>(p1, p2, pg, ln_g, ln_b,
                                                         relw_bf, rel_b, mask, out);
}
